// SOFTAttention_15702400434263
// MI455X (gfx1250) — matrix operand order repaired and checked by reading, not compiled, not run on hardware
//
#include <hip/hip_runtime.h>
#include <stdint.h>

// Softmax-free Gaussian-kernel attention for MI455X (gfx1250, wave32).
// O[b,h,i,:] = sum_j exp(-(||qi||^2+||qj||^2-2 qi.qj)/16) * V[b,h,j,:]
// GEMMs on v_wmma_f32_16x16x32_f16; K/V chunks streamed into LDS with
// double-buffered global_load_async_to_lds_b128 (ASYNCcnt).

typedef _Float16 v8h  __attribute__((ext_vector_type(8)));
typedef _Float16 v16h __attribute__((ext_vector_type(16)));
typedef float    v8f  __attribute__((ext_vector_type(8)));

union H16 { v16h v; v8h h[2]; };

__device__ __forceinline__ v8f wmma_f16(v16h a, v16h b, v8f c) {
  return __builtin_amdgcn_wmma_f32_16x16x32_f16(false, a, false, b,
                                                (short)0, c, false, false);
}

__device__ __forceinline__ v8h cvt8(const float* __restrict__ p) {
  float4 a = *(const float4*)p;
  float4 b = *(const float4*)(p + 4);
  v8h r;
  r[0] = (_Float16)a.x; r[1] = (_Float16)a.y; r[2] = (_Float16)a.z; r[3] = (_Float16)a.w;
  r[4] = (_Float16)b.x; r[5] = (_Float16)b.y; r[6] = (_Float16)b.z; r[7] = (_Float16)b.w;
  return r;
}

// One 16B async copy global->LDS per lane (GVS mode: sgpr base + vgpr offset).
__device__ __forceinline__ void async_copy16(uint32_t lds_dst, const void* sbase,
                                             uint32_t voff) {
  asm volatile("global_load_async_to_lds_b128 %0, %1, %2 offset:0"
               :: "v"(lds_dst), "v"(voff), "s"(sbase) : "memory");
}

// ---------------- prep pass 1: f32 -> f16 copies ----------------
// qh:  [BH][L][64] f16 row-major (A-fragments + K-chunk source, contiguous 4KB/chunk)
// vt:  [BH][L/32][64][32] f16 (V chunk pre-transposed: (d, k_local)) contiguous 4KB/chunk
__global__ __launch_bounds__(256) void prep_convert(
    const float* __restrict__ q, const float* __restrict__ v,
    _Float16* __restrict__ qh, _Float16* __restrict__ vt)
{
  const size_t e0 = ((size_t)blockIdx.x * blockDim.x + threadIdx.x) * 8;
  *(v8h*)(qh + e0) = cvt8(q + e0);

  const size_t row   = e0 >> 6;        // bh*L + i
  const int    d0    = (int)(e0 & 63);
  const size_t chunk = row >> 5;       // bh*(L/32) + i/32
  const int    k     = (int)(row & 31);
  float4 a = *(const float4*)(v + e0);
  float4 b = *(const float4*)(v + e0 + 4);
  _Float16* dst = vt + (chunk * 64 + (size_t)d0) * 32 + k;
  dst[0*32] = (_Float16)a.x; dst[1*32] = (_Float16)a.y;
  dst[2*32] = (_Float16)a.z; dst[3*32] = (_Float16)a.w;
  dst[4*32] = (_Float16)b.x; dst[5*32] = (_Float16)b.y;
  dst[6*32] = (_Float16)b.z; dst[7*32] = (_Float16)b.w;
}

// ---------------- prep pass 2: cq[r] = -||q_r||^2 * log2(e)/16 ----------------
__global__ __launch_bounds__(256) void prep_cq(
    const float* __restrict__ q, float* __restrict__ cq)
{
  const int r = blockIdx.x * blockDim.x + threadIdx.x;
  const float* p = q + (size_t)r * 64;
  float acc = 0.f;
#pragma unroll 8
  for (int i = 0; i < 64; ++i) acc += p[i] * p[i];
  cq[r] = acc * (-0.09016844005556021f);   // -log2(e)/16
}

// ---------------- main kernel ----------------
__global__ __launch_bounds__(256) void gauss_attn_wmma(
    const _Float16* __restrict__ qh, const _Float16* __restrict__ vt,
    const float* __restrict__ cq, float* __restrict__ out, int blocksPerSeq)
{
  constexpr int L = 2048, D = 64, NCHUNK = L / 32;
  constexpr int KJ_STRIDE = 72;   // halves per K row in LDS (64 + 8 pad -> 144B)
  constexpr int VT_STRIDE = 40;   // halves per V^T row in LDS (32 + 8 pad -> 80B)

  __shared__ __align__(16) _Float16 kjbuf[2][32 * KJ_STRIDE];
  __shared__ __align__(16) _Float16 vtbuf[2][64 * VT_STRIDE];
  __shared__ __align__(16) _Float16 s_lds[8 * 16 * 32];

  const int tid  = threadIdx.x;
  const int lane = tid & 31;
  const int wv   = tid >> 5;
  const int bh   = blockIdx.x / blocksPerSeq;
  const int row0 = (blockIdx.x % blocksPerSeq) * 128;

  const _Float16* qb  = qh + (size_t)bh * L * D;
  const _Float16* vtb = vt + (size_t)bh * NCHUNK * 64 * 32;
  const float*    cqb = cq + (size_t)bh * L;
  float*          ob  = out + (size_t)bh * L * D;

  const int n15  = lane & 15;           // A row / B,C column selector
  const int kb   = (lane >> 4) * 8;     // A-fragment K sub-offset
  const int kgrp = (lane >> 4) * 16;    // unused: B fragments also use kb

  // per-wave Q A-fragments straight from f16 global
  const _Float16* qrow = qb + (size_t)(row0 + wv * 16 + n15) * D;
  H16 aQ0, aQ1;
  aQ0.h[0] = *(const v8h*)(qrow + kb);       aQ0.h[1] = *(const v8h*)(qrow + 16 + kb);
  aQ1.h[0] = *(const v8h*)(qrow + 32 + kb);  aQ1.h[1] = *(const v8h*)(qrow + 48 + kb);

  // ci per C-slot: vgpr vv -> M = vv + 8*(lane>=16)
  float ci[8];
#pragma unroll
  for (int vv = 0; vv < 8; ++vv)
    ci[vv] = cqb[row0 + wv * 16 + ((lane >> 4) << 3) + vv];

  // async-copy lane geometry (each thread moves 16B of K and 16B of V per chunk)
  const int baseB = tid * 16;
  const int kRow = baseB >> 7, kW = baseB & 127;   // K chunk: 32 rows x 128B
  const int vRow = baseB >> 6, vW = baseB & 63;    // V chunk: 64 rows x 64B
  uint32_t kDst[2], vDst[2];
#pragma unroll
  for (int b = 0; b < 2; ++b) {
    kDst[b] = (uint32_t)(uintptr_t)&kjbuf[b][0] + (uint32_t)(kRow * 144 + kW);
    vDst[b] = (uint32_t)(uintptr_t)&vtbuf[b][0] + (uint32_t)(vRow * 80 + vW);
  }

  v8f Oacc[4] = {};

  // prologue: stream chunk 0 into buffer 0
  async_copy16(kDst[0], qb, (uint32_t)baseB);
  async_copy16(vDst[0], vtb, (uint32_t)baseB);

  for (int c = 0; c < NCHUNK; ++c) {
    const int j0 = c * 32;
    // prefetch next chunk (clamped; last iteration writes the idle buffer)
    const int cn = (c + 1 < NCHUNK) ? c + 1 : c;
    const int bn = (c + 1) & 1;
    async_copy16(kDst[bn], qb + (size_t)cn * 32 * D, (uint32_t)baseB);
    async_copy16(vDst[bn], vtb + (size_t)cn * 64 * 32, (uint32_t)baseB);

    // retire this chunk's (older) pair; next chunk's pair stays in flight
    asm volatile("s_wait_asynccnt 0x2" ::: "memory");
    __syncthreads();

    const int b = c & 1;
    const _Float16* kjb = &kjbuf[b][0];
    const _Float16* vjb = &vtbuf[b][0];

    // ---- dots = Q_i (16x64) x K_j^T : two 16-key tiles, K=64 -> 2 WMMAs each ----
    v8f dots[2];
#pragma unroll
    for (int t = 0; t < 2; ++t) {
      const _Float16* krow = kjb + (t * 16 + n15) * KJ_STRIDE;
      H16 b0, b1;
      b0.h[0] = *(const v8h*)(krow + kb);
      b0.h[1] = *(const v8h*)(krow + kb + 16);
      b1.h[0] = *(const v8h*)(krow + 32 + kb);
      b1.h[1] = *(const v8h*)(krow + 32 + kb + 16);
      v8f cc = {};
      cc = wmma_f16(aQ0.v, b0.v, cc);
      cc = wmma_f16(aQ1.v, b1.v, cc);
      dots[t] = cc;
    }

    // ---- S = exp2(d*log2e/8 + ci + cj), staged to wave-private LDS ----
    const float cj0 = cqb[j0 + n15];
    const float cj1 = cqb[j0 + 16 + n15];
    constexpr float K8 = 0.18033688011112042f;   // log2(e)/8
    _Float16* sbase = &s_lds[wv * 512];
#pragma unroll
    for (int t = 0; t < 2; ++t) {
      const float cj = t ? cj1 : cj0;
#pragma unroll
      for (int vv = 0; vv < 8; ++vv) {
        const float s = __builtin_amdgcn_exp2f(
            __builtin_fmaf(dots[t][vv], K8, ci[vv] + cj));
        const int M = vv + ((lane >> 4) << 3);
        sbase[M * 32 + t * 16 + n15] = (_Float16)s;
      }
    }

    // ---- O += S (16x32) x V_j (32x64) : 4 N-tiles, K=32 each ----
    H16 aS;
    const _Float16* srow = &s_lds[wv * 512 + n15 * 32];
    aS.h[0] = *(const v8h*)(srow + kb);
    aS.h[1] = *(const v8h*)(srow + 16 + kb);
#pragma unroll
    for (int t2 = 0; t2 < 4; ++t2) {
      const _Float16* vcol = vjb + (t2 * 16 + n15) * VT_STRIDE;
      H16 bV;
      bV.h[0] = *(const v8h*)(vcol + kb);
      bV.h[1] = *(const v8h*)(vcol + kb + 16);
      Oacc[t2] = wmma_f16(aS.v, bV.v, Oacc[t2]);
    }

    __syncthreads();   // reads of buf b done before it is prefetched again
  }

  // ---- write 16x64 output tile ----
#pragma unroll
  for (int t2 = 0; t2 < 4; ++t2) {
#pragma unroll
    for (int vv = 0; vv < 8; ++vv) {
      const int rr = row0 + wv * 16 + vv + ((lane >> 4) << 3);
      const int cc = t2 * 16 + n15;
      ob[(size_t)rr * D + cc] = Oacc[t2][vv];
    }
  }
}

extern "C" void kernel_launch(void* const* d_in, const int* in_sizes, int n_in,
                              void* d_out, int out_size, void* d_ws, size_t ws_size,
                              hipStream_t stream) {
  (void)n_in; (void)out_size; (void)ws_size;
  const float* q = (const float*)d_in[0];
  const float* v = (const float*)d_in[1];
  float* out = (float*)d_out;
  const int L = 2048, D = 64;
  const int BH = in_sizes[0] / (L * D);          // 16 for reference shapes
  const size_t nElem = (size_t)BH * L * D;       // 2,097,152

  // workspace layout: qh (f16), vt (f16), cq (f32) -> ~8.1 MB
  _Float16* qh = (_Float16*)d_ws;
  _Float16* vt = qh + nElem;
  float*    cq = (float*)(vt + nElem);

  prep_convert<<<dim3((unsigned)(nElem / 8 / 256)), dim3(256), 0, stream>>>(q, v, qh, vt);
  prep_cq<<<dim3((unsigned)(BH * L / 256)), dim3(256), 0, stream>>>(q, cq);

  const int blocksPerSeq = L / 128;               // 128 query rows per block (8 waves)
  gauss_attn_wmma<<<dim3(BH * blocksPerSeq), dim3(256), 0, stream>>>(qh, vt, cq, out, blocksPerSeq);
}
